// MultiheadAttention_1614907703649
// MI455X (gfx1250) — matrix operand order repaired and checked by reading, not compiled, not run on hardware
//
#include <hip/hip_runtime.h>

// ---------------------------------------------------------------------------
// MultiheadAttention on gfx1250 (MI455X).
// bf16 WMMA for all matmuls, async global->LDS staging, double-buffered GEMM.
// B=2, L=2048, D_MODEL=1024, H=16, D_HEAD=64
// ---------------------------------------------------------------------------

typedef __attribute__((ext_vector_type(16))) __bf16        v16bf;
typedef __attribute__((ext_vector_type(8)))  float         v8f;
typedef __attribute__((ext_vector_type(4)))  unsigned int  v4u;

union FragU { v16bf bf; v4u u[2]; };

static constexpr int BATCH = 2;
static constexpr int LSEQ  = 2048;
static constexpr int DM    = 1024;
static constexpr int NH    = 16;
static constexpr int DH    = 64;
static constexpr int MROWS = BATCH * LSEQ;   // 4096

static constexpr int BM = 128, BN = 128, BK = 32;
static constexpr int GPAD = 40;   // GEMM LDS row stride (ushorts): 80B = 5*16B, conflict-free
static constexpr int APAD = 72;   // attn LDS row stride: 144B = 9*16B, conflict-free

__device__ __forceinline__ unsigned short f2bf(float f) {
  unsigned int u = __float_as_uint(f);
  unsigned int r = u + 0x7FFFu + ((u >> 16) & 1u);   // round-to-nearest-even
  return (unsigned short)(r >> 16);
}

__device__ __forceinline__ v16bf frag_pair(const unsigned short* p, int off0, int off1) {
  FragU f;
  f.u[0] = *(const v4u*)(p + off0);
  f.u[1] = *(const v4u*)(p + off1);
  return f.bf;
}

// ---- CDNA5 async global->LDS copy (ASYNCcnt-tracked, bypasses VGPRs) ------
__device__ __forceinline__ unsigned lds_addr32(const void* p) {
  return (unsigned)(uintptr_t)p;   // low 32 bits of generic addr == LDS offset
}
__device__ __forceinline__ void async_cp16(unsigned lds, const void* g) {
  asm volatile("global_load_async_to_lds_b128 %0, %1, off"
               :: "v"(lds), "v"((unsigned long long)(uintptr_t)g) : "memory");
}
__device__ __forceinline__ void wait_async_le4() {
  asm volatile("s_wait_asynccnt 0x4" ::: "memory");
}
__device__ __forceinline__ void wait_async_0() {
  asm volatile("s_wait_asynccnt 0x0" ::: "memory");
}

// ---------------------------------------------------------------------------
// f32 -> bf16 flat convert (8 elements / thread)
// ---------------------------------------------------------------------------
__global__ __launch_bounds__(256)
void cvt_f32_to_bf16(const float* __restrict__ src, unsigned short* __restrict__ dst) {
  const size_t i = ((size_t)blockIdx.x * 256 + threadIdx.x) * 8;
  const float4 a = *(const float4*)(src + i);
  const float4 b = *(const float4*)(src + i + 4);
  v4u o;
  o.x = (unsigned)f2bf(a.x) | ((unsigned)f2bf(a.y) << 16);
  o.y = (unsigned)f2bf(a.z) | ((unsigned)f2bf(a.w) << 16);
  o.z = (unsigned)f2bf(b.x) | ((unsigned)f2bf(b.y) << 16);
  o.w = (unsigned)f2bf(b.z) | ((unsigned)f2bf(b.w) << 16);
  *(v4u*)(dst + i) = o;
}

// ---------------------------------------------------------------------------
// W[k][n] f32  ->  Wt[n][k] bf16   (32x32 tiles through LDS)
// ---------------------------------------------------------------------------
__global__ __launch_bounds__(256)
void transpose_w_bf16(const float* __restrict__ W, unsigned short* __restrict__ Wt) {
  __shared__ unsigned short t[32][33];
  const int tid = threadIdx.x;
  const int r  = tid >> 3;         // 0..31
  const int c  = (tid & 7) * 4;    // 0..28
  const int kb = blockIdx.y * 32;
  const int nb = blockIdx.x * 32;
  const float* s = W + (size_t)(kb + r) * DM + nb + c;
#pragma unroll
  for (int i = 0; i < 4; ++i) t[c + i][r] = f2bf(s[i]);
  __syncthreads();
  unsigned short* d = Wt + (size_t)(nb + r) * DM + kb + c;
#pragma unroll
  for (int i = 0; i < 4; ++i) d[i] = t[r][c + i];
}

// ---------------------------------------------------------------------------
// C[M x N] = A[M x K] * Wt[N x K]^T   (A, Wt bf16; double-buffered async LDS)
//   OUT_MODE 0: store bf16 into head layout [B][H][L][DH]   (QKV projections)
//   OUT_MODE 1: store f32 row-major [M][N]                  (output projection)
// ---------------------------------------------------------------------------
template <int OUT_MODE>
__global__ __launch_bounds__(256)
void gemm_bf16_wmma(const unsigned short* __restrict__ A,
                    const unsigned short* __restrict__ Bt,
                    void* __restrict__ Cptr, int M, int N, int K)
{
  __shared__ alignas(16) unsigned short As[2][BM * GPAD];   // [m][k]
  __shared__ alignas(16) unsigned short Bs[2][BN * GPAD];   // [n][k]

  const int tid  = threadIdx.x;
  const int lane = tid & 31;
  const int wave = tid >> 5;
  const int wm   = wave & 3;
  const int wn   = wave >> 2;
  const int m0   = blockIdx.y * BM;
  const int n0   = blockIdx.x * BN;

  const int arow = lane & 15;
  const int half = lane >> 4;
  const int aoff = half << 3;
  const int bko  = half << 4;

  const int srow  = tid >> 2;       // 0..63
  const int spart = tid & 3;        // 4 x 16B = 64B = 32 bf16 = BK

  v8f acc[2][4] = {};

  auto stage = [&](int buf, int kk) {
    const unsigned short* ga = A  + (size_t)(m0 + srow) * K + kk + spart * 8;
    const unsigned short* gb = Bt + (size_t)(n0 + srow) * K + kk + spart * 8;
    const unsigned la = lds_addr32(&As[buf][srow * GPAD]) + spart * 16;
    const unsigned lb = lds_addr32(&Bs[buf][srow * GPAD]) + spart * 16;
    async_cp16(la,                   ga);
    async_cp16(la + 64 * GPAD * 2,   ga + (size_t)64 * K);
    async_cp16(lb,                   gb);
    async_cp16(lb + 64 * GPAD * 2,   gb + (size_t)64 * K);
  };

  stage(0, 0);
  const int NIT = K / BK;
  for (int it = 0; it < NIT; ++it) {
    const int p = it & 1;
    if (it + 1 < NIT) {
      stage(1 - p, (it + 1) * BK);   // prefetch next tile into other buffer
      wait_async_le4();              // oldest 4 (current buffer) have landed
    } else {
      wait_async_0();
    }
    __syncthreads();

    const unsigned short* abase = &As[p][(wm * 32) * GPAD];
    const unsigned short* bbase = &Bs[p][(wn * 64) * GPAD];
#pragma unroll
    for (int mi = 0; mi < 2; ++mi) {
      v16bf a = frag_pair(abase + (mi * 16 + arow) * GPAD, aoff, 16 + aoff);
#pragma unroll
      for (int ni = 0; ni < 4; ++ni) {
        v16bf b = frag_pair(bbase + (ni * 16 + arow) * GPAD, aoff, 16 + aoff);
        acc[mi][ni] = __builtin_amdgcn_wmma_f32_16x16x32_bf16(
            false, a, false, b, (short)0, acc[mi][ni], false, false);
      }
    }
    __syncthreads();
  }

  const int mrow = half << 3;
#pragma unroll
  for (int mi = 0; mi < 2; ++mi) {
#pragma unroll
    for (int ni = 0; ni < 4; ++ni) {
#pragma unroll
      for (int r = 0; r < 8; ++r) {
        const int gm = m0 + wm * 32 + mi * 16 + mrow + r;
        const int gn = n0 + wn * 64 + ni * 16 + arow;
        if (OUT_MODE == 0) {
          const int b = gm >> 11, q = gm & (LSEQ - 1);
          const int h = gn >> 6,  d = gn & (DH - 1);
          ((unsigned short*)Cptr)[(((size_t)b * NH + h) * LSEQ + q) * DH + d] =
              f2bf(acc[mi][ni][r]);
        } else {
          ((float*)Cptr)[(size_t)gm * N + gn] = acc[mi][ni][r];
        }
      }
    }
  }
}

// ---------------------------------------------------------------------------
// Causal flash attention per (b, h): softmax(QK^T masked) * V, scaled by 1/8
// (reference applies softmax first, then divides by sqrt(D_HEAD)=8).
// Block = 128 q-rows (8 waves x 16 rows); 64-key tiles; K staged async.
// ---------------------------------------------------------------------------
__global__ __launch_bounds__(256)
void attn_wmma(const unsigned short* __restrict__ Qh,
               const unsigned short* __restrict__ Kh,
               const unsigned short* __restrict__ Vh,
               unsigned short* __restrict__ Att)
{
  __shared__ alignas(16) unsigned short Ks[64 * APAD];       // [key][d]
  __shared__ alignas(16) unsigned short Vs[64 * APAD];       // [d][key]
  __shared__ alignas(16) unsigned short Ps[8][16 * APAD];    // per-wave P tile

  const int tid  = threadIdx.x;
  const int lane = tid & 31;
  const int wave = tid >> 5;
  const int q0   = blockIdx.x * 128;
  const int b    = blockIdx.y >> 4;
  const int h    = blockIdx.y & (NH - 1);

  const size_t headoff = ((size_t)b * NH + h) * LSEQ * DH;
  const unsigned short* Qp = Qh + headoff;
  const unsigned short* Kp = Kh + headoff;
  const unsigned short* Vp = Vh + headoff;

  const int qw   = q0 + wave * 16;
  const int arow = lane & 15;
  const int half = lane >> 4;
  const int aoff = half << 3;
  const int bko  = half << 4;

  v16bf qf[2];
  {
    const unsigned short* qrow = Qp + (size_t)(qw + arow) * DH;
#pragma unroll
    for (int s = 0; s < 2; ++s)
      qf[s] = frag_pair(qrow + s * 32, aoff, 16 + aoff);
  }

  v8f   o[4] = {};
  float mrow[8], lrow[8];
#pragma unroll
  for (int r = 0; r < 8; ++r) { mrow[r] = -3.0e38f; lrow[r] = 0.f; }

  for (int kb = 0; kb < q0 + 128; kb += 64) {
    __syncthreads();   // previous tile fully consumed by all waves
    {
      // K tile: async copy (layout-identical), 2 x 16B chunks per thread
      const int krow  = tid >> 3;       // 0..31
      const int kpart = tid & 7;        // 8 x 16B = 128B per row
      const unsigned short* gk = Kp + (size_t)(kb + krow) * DH + kpart * 8;
      const unsigned lk = lds_addr32(&Ks[krow * APAD]) + kpart * 16;
      async_cp16(lk,                  gk);
      async_cp16(lk + 32 * APAD * 2,  gk + (size_t)32 * DH);

      // V tile: transposed into LDS (manual path, overlaps the async copy)
      const int vrow = tid >> 2;        // key 0..63
      const int vcb  = (tid & 3) * 16;  // d sub-block
      const unsigned short* vsrc = Vp + (size_t)(kb + vrow) * DH + vcb;
#pragma unroll
      for (int i = 0; i < 16; ++i)
        Vs[(vcb + i) * APAD + vrow] = vsrc[i];
      if (kb + 64 < q0 + 128)
        __builtin_prefetch(Vp + (size_t)(kb + 64 + vrow) * DH + vcb, 0, 0);
      wait_async_0();
    }
    __syncthreads();

    if (kb <= qw + 15) {   // uniform per wave: skip fully-masked key blocks
      // ---- S = Q K^T (16 x 64) ----
      v8f sacc[4] = {};
#pragma unroll
      for (int ni = 0; ni < 4; ++ni) {
#pragma unroll
        for (int s = 0; s < 2; ++s) {
          v16bf bf = frag_pair(&Ks[(ni * 16 + arow) * APAD + s * 32], aoff, 16 + aoff);
          sacc[ni] = __builtin_amdgcn_wmma_f32_16x16x32_bf16(
              false, qf[s], false, bf, (short)0, sacc[ni], false, false);
        }
      }

      // ---- causal mask + online softmax ----
#pragma unroll
      for (int r = 0; r < 8; ++r) {
        const int qi = qw + (half << 3) + r;
        float rmax = -3.0e38f;
#pragma unroll
        for (int ni = 0; ni < 4; ++ni) {
          const int ki = kb + ni * 16 + arow;
          float sv = (ki > qi) ? -1.0e9f : sacc[ni][r];
          sacc[ni][r] = sv;
          rmax = fmaxf(rmax, sv);
        }
#pragma unroll
        for (int off = 8; off >= 1; off >>= 1)
          rmax = fmaxf(rmax, __shfl_xor(rmax, off, 32));
        const float mnew  = fmaxf(mrow[r], rmax);
        const float alpha = __expf(mrow[r] - mnew);
        mrow[r] = mnew;
        float rsum = 0.f;
#pragma unroll
        for (int ni = 0; ni < 4; ++ni) {
          const float p = __expf(sacc[ni][r] - mnew);
          sacc[ni][r] = p;
          rsum += p;
        }
#pragma unroll
        for (int off = 8; off >= 1; off >>= 1)
          rsum += __shfl_xor(rsum, off, 32);
        lrow[r] = lrow[r] * alpha + rsum;
#pragma unroll
        for (int dt = 0; dt < 4; ++dt) o[dt][r] *= alpha;
      }

      // ---- stage P (C-layout -> A-layout) through per-wave LDS ----
      unsigned short* pw = &Ps[wave][0];
#pragma unroll
      for (int ni = 0; ni < 4; ++ni)
#pragma unroll
        for (int r = 0; r < 8; ++r)
          pw[((half << 3) + r) * APAD + ni * 16 + arow] = f2bf(sacc[ni][r]);

      // ---- O += P * V ----
#pragma unroll
      for (int dt = 0; dt < 4; ++dt) {
#pragma unroll
        for (int s = 0; s < 2; ++s) {
          v16bf pa = frag_pair(pw + arow * APAD + s * 32, aoff, 16 + aoff);
          v16bf vb = frag_pair(&Vs[(dt * 16 + arow) * APAD + s * 32], aoff, 16 + aoff);
          o[dt] = __builtin_amdgcn_wmma_f32_16x16x32_bf16(
              false, pa, false, vb, (short)0, o[dt], false, false);
        }
      }
    }
  }

  // ---- normalize (1/l) * 1/sqrt(DH), store bf16 into [B][L][DM] ----
#pragma unroll
  for (int r = 0; r < 8; ++r) {
    const int qi  = qw + (half << 3) + r;
    const float s = 0.125f / lrow[r];
#pragma unroll
    for (int dt = 0; dt < 4; ++dt) {
      const int col = h * DH + dt * 16 + arow;
      Att[((size_t)b * LSEQ + qi) * DM + col] = f2bf(o[dt][r] * s);
    }
  }
}

// ---------------------------------------------------------------------------
extern "C" void kernel_launch(void* const* d_in, const int* in_sizes, int n_in,
                              void* d_out, int out_size, void* d_ws, size_t ws_size,
                              hipStream_t stream) {
  const float* q  = (const float*)d_in[0];
  const float* k  = (const float*)d_in[1];
  const float* v  = (const float*)d_in[2];
  // d_in[3] = mask (causal; computed analytically in-kernel)
  const float* Wq = (const float*)d_in[4];
  const float* Wk = (const float*)d_in[5];
  const float* Wv = (const float*)d_in[6];
  const float* Wo = (const float*)d_in[7];
  float* out = (float*)d_out;

  // workspace (bf16 ushorts), 40 MB total:
  //   buf1: 8MB staging (cycled: q/k/v bf16, then attention output)
  //   Wqt..Wot: 4 x 2MB transposed weights
  //   Qh/Kh/Vh: 3 x 8MB head-layout projections
  const size_t seg  = (size_t)MROWS * DM;   // 4,194,304
  const size_t wseg = (size_t)DM * DM;      // 1,048,576
  unsigned short* buf1 = (unsigned short*)d_ws;
  unsigned short* Wqt  = buf1 + seg;
  unsigned short* Wkt  = Wqt + wseg;
  unsigned short* Wvt  = Wkt + wseg;
  unsigned short* Wot  = Wvt + wseg;
  unsigned short* Qhw  = Wot + wseg;
  unsigned short* Khw  = Qhw + seg;
  unsigned short* Vhw  = Khw + seg;

  dim3 blk(256);
  dim3 tg(DM / 32, DM / 32);               // (32, 32)
  transpose_w_bf16<<<tg, blk, 0, stream>>>(Wq, Wqt);
  transpose_w_bf16<<<tg, blk, 0, stream>>>(Wk, Wkt);
  transpose_w_bf16<<<tg, blk, 0, stream>>>(Wv, Wvt);
  transpose_w_bf16<<<tg, blk, 0, stream>>>(Wo, Wot);

  const int cvtblocks = (int)(seg / 8 / 256);   // 2048
  dim3 gg(DM / BN, MROWS / BM);                 // (8, 32)

  cvt_f32_to_bf16<<<cvtblocks, blk, 0, stream>>>(q, buf1);
  gemm_bf16_wmma<0><<<gg, blk, 0, stream>>>(buf1, Wqt, Qhw, MROWS, DM, DM);
  cvt_f32_to_bf16<<<cvtblocks, blk, 0, stream>>>(k, buf1);
  gemm_bf16_wmma<0><<<gg, blk, 0, stream>>>(buf1, Wkt, Khw, MROWS, DM, DM);
  cvt_f32_to_bf16<<<cvtblocks, blk, 0, stream>>>(v, buf1);
  gemm_bf16_wmma<0><<<gg, blk, 0, stream>>>(buf1, Wvt, Vhw, MROWS, DM, DM);

  dim3 ga(LSEQ / 128, BATCH * NH);              // (16, 32)
  attn_wmma<<<ga, blk, 0, stream>>>(Qhw, Khw, Vhw, buf1);

  gemm_bf16_wmma<1><<<gg, blk, 0, stream>>>(buf1, Wot, out, MROWS, DM, DM);
}
